// PreT_Attention_11020886081857
// MI455X (gfx1250) — matrix operand order repaired and checked by reading, not compiled, not run on hardware
//
#include <hip/hip_runtime.h>
#include <hip/hip_bf16.h>

// ---------------------------------------------------------------------------
// BEiT attention block for MI455X (gfx1250, wave32, WMMA).
// ~134 GFLOP vs ~7us of HBM traffic at 23.3 TB/s -> compute-bound, so all
// three GEMM stages run on v_wmma_f32_16x16x32_bf16 (fp32 accumulate).
// Weights down-converted to bf16 once; relative-position bias pre-expanded to
// rpbx[h][qi][224] with -3e38 padding so the attention inner loop has no
// dependent gather chain or masking branches.
// ---------------------------------------------------------------------------

#define DIM_  768
#define H_    12
#define HD_   64
#define B_    128
#define N_    197
#define NKP_  224                // padded key count (7 chunks of 32)
#define BN_   (B_ * N_)          // 25216 = 1576 * 16, exact

typedef __bf16 v16bf __attribute__((ext_vector_type(16)));
typedef __bf16 v4bf  __attribute__((ext_vector_type(4)));
typedef float  v8f   __attribute__((ext_vector_type(8)));
typedef unsigned int u32x8 __attribute__((ext_vector_type(8)));

__device__ __forceinline__ __bf16 f2bf(float f) { return (__bf16)f; }

// A-matrix (16x32 bf16): row M = lane&15; lane<16 holds K=0..15, lane>=16 K=16..31
__device__ __forceinline__ int a_kidx(int lane, int h) {
    return h + ((lane & 16) ? 16 : 0);
}
// B-matrix (32x16 bf16): col N = lane&15; lane<16 holds K=0..15, lane>=16 K=16..31
__device__ __forceinline__ int b_kidx(int lane, int h) {
    return h + ((lane & 16) ? 16 : 0);
}

// ---------------------------------------------------------------------------
// Kernel 0a: bulk f32 -> bf16 conversion (weights; run once, tiny)
// ---------------------------------------------------------------------------
__global__ __launch_bounds__(256)
void cvt_bf16_kernel(const float* __restrict__ in, __bf16* __restrict__ out, int n4) {
    int i = blockIdx.x * 256 + threadIdx.x;
    if (i >= n4) return;
    float4 f = ((const float4*)in)[i];
    v4bf o;
    o.x = f2bf(f.x); o.y = f2bf(f.y); o.z = f2bf(f.z); o.w = f2bf(f.w);
    ((v4bf*)out)[i] = o;
}

// ---------------------------------------------------------------------------
// Kernel 0b: expand rel-pos bias: rpbx[h][qi][key] = rpb_table[rel_idx[qi,key],h]
// padded keys (>=197) pre-filled with -3e38 (softmax mask baked in).
// ---------------------------------------------------------------------------
__global__ __launch_bounds__(256)
void rpb_expand_kernel(const float* __restrict__ rpb, const int* __restrict__ rel_idx,
                       float* __restrict__ rpbx) {
    int i = blockIdx.x * 256 + threadIdx.x;       // over H_*N_*NKP_
    if (i >= H_ * N_ * NKP_) return;
    int key  = i % NKP_;
    int rest = i / NKP_;
    int qi   = rest % N_;
    int h    = rest / N_;
    float v = -3.0e38f;
    if (key < N_) v = rpb[rel_idx[qi * N_ + key] * H_ + h];
    rpbx[i] = v;
}

// ---------------------------------------------------------------------------
// Kernel 1: qkv = x @ qkv_w^T + [q_bias, 0, v_bias]; q *= HD^-0.5
// Writes q,k,v as bf16 in [B, H, N, HD]. 4 waves; wave tile 16x96 (6 WMMA
// per 32-wide K-step sharing one A fragment).
// ---------------------------------------------------------------------------
__global__ __launch_bounds__(128)
void qkv_proj_kernel(const float* __restrict__ x, const __bf16* __restrict__ wq,
                     const float* __restrict__ q_bias, const float* __restrict__ v_bias,
                     __bf16* __restrict__ qo, __bf16* __restrict__ ko, __bf16* __restrict__ vo) {
    __shared__ __bf16 lds_a[16][32];

    const int tid  = threadIdx.x;
    const int lane = tid & 31;
    const int wave = tid >> 5;
    const int mrow0 = blockIdx.x * 16;
    const int col0  = blockIdx.y * 384 + wave * 96;
    const int nn    = lane & 15;
    const int bhalf = (lane & 16) ? 16 : 0;

    v8f acc[6] = {};

    for (int kt = 0; kt < DIM_; kt += 32) {
        __syncthreads();
        {   // stage 16x32 activation tile (fp32 -> bf16) cooperatively
            int idx = tid * 4;                   // 512 elements / 128 threads
            int m = idx >> 5, kk = idx & 31;
            const float* src = x + (size_t)(mrow0 + m) * DIM_ + kt + kk;
            if (kt + 32 < DIM_) __builtin_prefetch(src + 32, 0, 3);
            lds_a[m][kk + 0] = f2bf(src[0]);
            lds_a[m][kk + 1] = f2bf(src[1]);
            lds_a[m][kk + 2] = f2bf(src[2]);
            lds_a[m][kk + 3] = f2bf(src[3]);
        }
        __syncthreads();

        v16bf a;
#pragma unroll
        for (int h = 0; h < 16; ++h) a[h] = lds_a[nn][a_kidx(lane, h)];

#pragma unroll
        for (int t = 0; t < 6; ++t) {
            const __bf16* wrow = wq + (size_t)(col0 + t * 16 + nn) * DIM_ + kt + bhalf;
            v16bf bfrag = __builtin_bit_cast(v16bf, *(const u32x8*)wrow);
            acc[t] = __builtin_amdgcn_wmma_f32_16x16x32_bf16(
                false, a, false, bfrag, (short)0, acc[t], false, false);
        }
    }

    const float scale = 0.125f;  // HD^-0.5
#pragma unroll
    for (int t = 0; t < 6; ++t) {
        int col   = col0 + t * 16 + nn;     // 0..2303
        int which = col / DIM_;             // 0=q 1=k 2=v
        int cm    = col % DIM_;
        int head  = cm / HD_, hd = cm % HD_;
        float bias = (which == 0) ? q_bias[cm] : ((which == 2) ? v_bias[cm] : 0.0f);
        __bf16* dst = (which == 0) ? qo : ((which == 1) ? ko : vo);
#pragma unroll
        for (int i = 0; i < 8; ++i) {
            int m = i + ((lane & 16) ? 8 : 0);
            int r = mrow0 + m;
            int bb = r / N_, tok = r % N_;
            float val = acc[t][i] + bias;
            if (which == 0) val *= scale;
            dst[((size_t)(bb * H_ + head) * N_ + tok) * HD_ + hd] = f2bf(val);
        }
    }
}

// ---------------------------------------------------------------------------
// Kernel 2: flash attention. Block = 64 threads (2 waves), each wave owns one
// 16-query tile; the two waves SHARE the staged 32x64 K/V tiles (halves the
// K/V traffic). Keys in 7 chunks of 32 (padded 224; mask baked into rpbx).
// ---------------------------------------------------------------------------
__global__ __launch_bounds__(64)
void attn_kernel(const __bf16* __restrict__ q, const __bf16* __restrict__ k,
                 const __bf16* __restrict__ v, const float* __restrict__ rpbx,
                 __bf16* __restrict__ ao) {
    __shared__ __bf16 lds_q[2][16 * HD_];
    __shared__ __bf16 lds_k[32 * HD_];
    __shared__ __bf16 lds_v[32 * HD_];
    __shared__ float  lds_p[2][16 * 32];

    const int tid  = threadIdx.x;
    const int lane = tid & 31;
    const int wv   = tid >> 5;
    const int qt = blockIdx.x * 2 + wv;       // 0..13 (tile 13 fully masked)
    const int h = blockIdx.y, b = blockIdx.z;
    const size_t bh = (size_t)(b * H_ + h) * N_ * HD_;
    const __bf16* qb = q + bh;
    const __bf16* kb = k + bh;
    const __bf16* vb = v + bh;
    const float* rpbh = rpbx + (size_t)h * N_ * NKP_;

    // stage this wave's Q tile (16x64 bf16)
#pragma unroll
    for (int it = 0; it < 4; ++it) {
        int u = lane + it * 32;              // 0..127 uint4 slots
        int m = u >> 3, c8 = (u & 7) * 8;
        int qi = qt * 16 + m;
        uint4 val = make_uint4(0, 0, 0, 0);
        if (qi < N_) val = *(const uint4*)(qb + (size_t)qi * HD_ + c8);
        *(uint4*)(&lds_q[wv][m * HD_ + c8]) = val;
    }
    __syncthreads();

    const int nn = lane & 15;
    const int hi8 = (lane & 16) ? 8 : 0;

    v16bf aq0, aq1;   // Q fragments: channels 0..31 and 32..63
#pragma unroll
    for (int hh = 0; hh < 16; ++hh) {
        int kk = a_kidx(lane, hh);
        aq0[hh] = lds_q[wv][nn * HD_ + kk];
        aq1[hh] = lds_q[wv][nn * HD_ + 32 + kk];
    }

    v8f oacc[4] = {};
    float mrow[8], lrow[8];
#pragma unroll
    for (int i = 0; i < 8; ++i) { mrow[i] = -3.0e38f; lrow[i] = 0.0f; }

    for (int jt = 0; jt < 7; ++jt) {
        const int kb0 = jt * 32;
        __syncthreads();
        // cooperative staging of shared 32x64 K and V tiles (64 threads)
#pragma unroll
        for (int it = 0; it < 4; ++it) {
            int u = tid + it * 64;           // 0..255 uint4 slots
            int row = u >> 3, c8 = (u & 7) * 8;
            int key = kb0 + row;
            uint4 kv4 = make_uint4(0, 0, 0, 0), vv4 = make_uint4(0, 0, 0, 0);
            if (key < N_) {
                kv4 = *(const uint4*)(kb + (size_t)key * HD_ + c8);
                vv4 = *(const uint4*)(vb + (size_t)key * HD_ + c8);
            }
            *(uint4*)(lds_k + row * HD_ + c8) = kv4;
            *(uint4*)(lds_v + row * HD_ + c8) = vv4;
        }
        __syncthreads();

        // S0 = Q @ K[0..15]^T, S1 = Q @ K[16..31]^T  (2 WMMA each over HD=64)
        v8f s0 = {}, s1 = {};
        {
            v16bf bk;
#pragma unroll
            for (int hh = 0; hh < 16; ++hh) bk[hh] = lds_k[nn * HD_ + b_kidx(lane, hh)];
            s0 = __builtin_amdgcn_wmma_f32_16x16x32_bf16(false, aq0, false, bk, (short)0, s0, false, false);
#pragma unroll
            for (int hh = 0; hh < 16; ++hh) bk[hh] = lds_k[nn * HD_ + 32 + b_kidx(lane, hh)];
            s0 = __builtin_amdgcn_wmma_f32_16x16x32_bf16(false, aq1, false, bk, (short)0, s0, false, false);
#pragma unroll
            for (int hh = 0; hh < 16; ++hh) bk[hh] = lds_k[(16 + nn) * HD_ + b_kidx(lane, hh)];
            s1 = __builtin_amdgcn_wmma_f32_16x16x32_bf16(false, aq0, false, bk, (short)0, s1, false, false);
#pragma unroll
            for (int hh = 0; hh < 16; ++hh) bk[hh] = lds_k[(16 + nn) * HD_ + 32 + b_kidx(lane, hh)];
            s1 = __builtin_amdgcn_wmma_f32_16x16x32_bf16(false, aq1, false, bk, (short)0, s1, false, false);
        }

        // + pre-expanded bias (mask baked in), online softmax update
#pragma unroll
        for (int i = 0; i < 8; ++i) {
            int m  = i + hi8;
            int qi = qt * 16 + m;
            int qiL = (qi < N_) ? qi : (N_ - 1);          // clamp garbage rows
            const float* rr = rpbh + (size_t)qiL * NKP_ + kb0 + nn;
            float sv0 = s0[i] + rr[0];
            float sv1 = s1[i] + rr[16];

            float t = fmaxf(sv0, sv1);                 // row-max across 16 lanes
            t = fmaxf(t, __shfl_xor(t, 1));
            t = fmaxf(t, __shfl_xor(t, 2));
            t = fmaxf(t, __shfl_xor(t, 4));
            t = fmaxf(t, __shfl_xor(t, 8));
            float mnew = fmaxf(mrow[i], t);
            float corr = __expf(mrow[i] - mnew);
            mrow[i] = mnew;

            float p0 = __expf(sv0 - mnew);
            float p1 = __expf(sv1 - mnew);
            float rs = p0 + p1;                        // row-sum across 16 lanes
            rs += __shfl_xor(rs, 1);
            rs += __shfl_xor(rs, 2);
            rs += __shfl_xor(rs, 4);
            rs += __shfl_xor(rs, 8);
            lrow[i] = lrow[i] * corr + rs;
            oacc[0][i] *= corr; oacc[1][i] *= corr;
            oacc[2][i] *= corr; oacc[3][i] *= corr;

            lds_p[wv][m * 32 + nn]      = p0;          // C-layout -> LDS
            lds_p[wv][m * 32 + 16 + nn] = p1;
        }

        // P (16x32) as A fragment; V chunk as B; accumulate O (16x64)
        v16bf ap;
#pragma unroll
        for (int hh = 0; hh < 16; ++hh) ap[hh] = f2bf(lds_p[wv][nn * 32 + a_kidx(lane, hh)]);
#pragma unroll
        for (int c = 0; c < 4; ++c) {
            v16bf bv;
#pragma unroll
            for (int hh = 0; hh < 16; ++hh)
                bv[hh] = lds_v[b_kidx(lane, hh) * HD_ + c * 16 + nn];
            oacc[c] = __builtin_amdgcn_wmma_f32_16x16x32_bf16(
                false, ap, false, bv, (short)0, oacc[c], false, false);
        }
    }

    // normalize and store bf16 [B, N, H*HD]
#pragma unroll
    for (int i = 0; i < 8; ++i) {
        int m  = i + hi8;
        int qi = qt * 16 + m;
        if (qi >= N_) continue;
        float inv = 1.0f / lrow[i];
        size_t base = ((size_t)b * N_ + qi) * DIM_ + h * HD_;
#pragma unroll
        for (int c = 0; c < 4; ++c)
            ao[base + c * 16 + nn] = f2bf(oacc[c][i] * inv);
    }
}

// ---------------------------------------------------------------------------
// Kernel 3: out = attn @ proj_w^T + proj_b  (fp32 output); wave tile 16x96.
// ---------------------------------------------------------------------------
__global__ __launch_bounds__(128)
void proj_kernel(const __bf16* __restrict__ attn, const __bf16* __restrict__ wp,
                 const float* __restrict__ proj_b, float* __restrict__ out) {
    __shared__ __bf16 lds_a[16][32];

    const int tid  = threadIdx.x;
    const int lane = tid & 31;
    const int wave = tid >> 5;
    const int mrow0 = blockIdx.x * 16;
    const int col0  = blockIdx.y * 384 + wave * 96;
    const int nn    = lane & 15;
    const int bhalf = (lane & 16) ? 16 : 0;

    v8f acc[6] = {};

    for (int kt = 0; kt < DIM_; kt += 32) {
        __syncthreads();
        {   // stage 16x32 bf16 activation tile (8-byte vector copies)
            int idx = tid * 4;
            int m = idx >> 5, kk = idx & 31;
            const __bf16* src = attn + (size_t)(mrow0 + m) * DIM_ + kt + kk;
            if (kt + 32 < DIM_) __builtin_prefetch(src + 32, 0, 3);
            *(uint2*)(&lds_a[m][kk]) = *(const uint2*)src;
        }
        __syncthreads();

        v16bf a;
#pragma unroll
        for (int hh = 0; hh < 16; ++hh) a[hh] = lds_a[nn][a_kidx(lane, hh)];

#pragma unroll
        for (int t = 0; t < 6; ++t) {
            const __bf16* wrow = wp + (size_t)(col0 + t * 16 + nn) * DIM_ + kt + bhalf;
            v16bf bfrag = __builtin_bit_cast(v16bf, *(const u32x8*)wrow);
            acc[t] = __builtin_amdgcn_wmma_f32_16x16x32_bf16(
                false, a, false, bfrag, (short)0, acc[t], false, false);
        }
    }

#pragma unroll
    for (int t = 0; t < 6; ++t) {
        int col = col0 + t * 16 + nn;
        float bias = proj_b[col];
#pragma unroll
        for (int i = 0; i < 8; ++i) {
            int m = i + ((lane & 16) ? 8 : 0);
            out[(size_t)(mrow0 + m) * DIM_ + col] = acc[t][i] + bias;
        }
    }
}

// ---------------------------------------------------------------------------
extern "C" void kernel_launch(void* const* d_in, const int* in_sizes, int n_in,
                              void* d_out, int out_size, void* d_ws, size_t ws_size,
                              hipStream_t stream) {
    const float* x      = (const float*)d_in[0];
    const float* qkv_w  = (const float*)d_in[1];
    const float* q_bias = (const float*)d_in[2];
    const float* v_bias = (const float*)d_in[3];
    const float* rpb    = (const float*)d_in[4];
    const float* proj_w = (const float*)d_in[5];
    const float* proj_b = (const float*)d_in[6];
    const int*   rel_i  = (const int*)d_in[7];
    float* out = (float*)d_out;

    const size_t QKV    = (size_t)B_ * H_ * N_ * HD_;  // 19,365,888 elements
    const int    NWQKV  = 3 * DIM_ * DIM_;             // 1,769,472
    const int    NWPROJ = DIM_ * DIM_;                 // 589,824
    const int    NRPBX  = H_ * N_ * NKP_;              // 529,536

    __bf16* wqb  = (__bf16*)d_ws;                      // bf16 qkv weights
    __bf16* wpb  = wqb + NWQKV;                        // bf16 proj weights
    float*  rpbx = (float*)(wpb + NWPROJ);             // expanded bias (f32)
    __bf16* qo   = (__bf16*)(rpbx + NRPBX);
    __bf16* ko   = qo + QKV;
    __bf16* vo   = ko + QKV;
    __bf16* ao   = vo + QKV;                           // total ~162 MB

    cvt_bf16_kernel<<<dim3((NWQKV / 4 + 255) / 256), 256, 0, stream>>>(qkv_w, wqb, NWQKV / 4);
    cvt_bf16_kernel<<<dim3((NWPROJ / 4 + 255) / 256), 256, 0, stream>>>(proj_w, wpb, NWPROJ / 4);
    rpb_expand_kernel<<<dim3((NRPBX + 255) / 256), 256, 0, stream>>>(rpb, rel_i, rpbx);

    qkv_proj_kernel<<<dim3(BN_ / 16, 6), 128, 0, stream>>>(x, wqb, q_bias, v_bias, qo, ko, vo);
    attn_kernel<<<dim3(7, H_, B_), 64, 0, stream>>>(qo, ko, vo, rpbx, ao);
    proj_kernel<<<dim3(BN_ / 16, 2), 128, 0, stream>>>(ao, wpb, proj_b, out);
}
